// FiLMModel_54778012893375
// MI455X (gfx1250) — matrix operand order repaired and checked by reading, not compiled, not run on hardware
//
#include <hip/hip_runtime.h>
#include <hip/hip_bf16.h>

// ---------------------------------------------------------------------------
// FiLM model, algebraically collapsed:
//   k_norm       : RevIN stats + normalized f16 signal Fh (192 x 1024)
//   k_ebar       : mean over L of eval_last                 (3 x 256)
//   k_wconv      : w_spec -> f16, [s][k][o][i] (K-contiguous B layout)
//   k_precompute : sequential A^d B sweep -> G (f16, column-major) per scale
//   k_gemm1      : X = Fh @ G   (WMMA f16; A panel staged in LDS via
//                  global_load_async_to_lds_b128 / ASYNCcnt; 4 N-tiles/wave)
//   k_gemm2      : complex mode-mix + gamma_k phase reduction (WMMA f16;
//                  B slices DMA'd by the Tensor Data Mover, double-buffered
//                  in LDS with s_wait_tensorcnt)
//   k_final      : decode-mean, scale-MLP, de-norm, mean, FC head
// ---------------------------------------------------------------------------

#define B_    16
#define L_    1024
#define C_    12
#define NH    256
#define MODES 32
#define NSC   3
#define BC    (B_ * C_)          // 192  (12 M-tiles of 16)
#define NCOL  (NH * 2 * MODES)   // 16384 columns of G
#define OUTD  5

typedef __attribute__((ext_vector_type(16))) _Float16 v16h;
typedef __attribute__((ext_vector_type(8)))  float    v8f;
typedef __attribute__((ext_vector_type(4)))  unsigned u32x4;
typedef __attribute__((ext_vector_type(8)))  unsigned u32x8;

__device__ __forceinline__ void put2(v16h& a, int idx, unsigned w) {
  union { unsigned u; _Float16 h[2]; } c; c.u = w;
  a[idx] = c.h[0]; a[idx + 1] = c.h[1];
}

__device__ __forceinline__ v8f wmma16(v16h a, v16h b, v8f c) {
  // v_wmma_f32_16x16x32_f16  D = A(16x32) * B(32x16) + C
  return __builtin_amdgcn_wmma_f32_16x16x32_f16(
      false, a, false, b, (short)0, c, false, false);
}

// A-fragment (16x32 f16, ISA 7.12.2): lane m=lane&15, g=lane>>4.
__device__ __forceinline__ int kA0(int v, int g) {
  return ((v < 4) ? 2 * v : 16 + 2 * (v - 4)) + 8 * g;
}
// B-fragment (32x16 f16): lane n=lane&15, g=lane>>4.
__device__ __forceinline__ int kB0(int v, int g) { return kA0(v, g); }

// Tensor Data Mover: 1D contiguous DMA of `nElem8` 8-byte elements from
// global to LDS.  D# per CDNA5 ISA: group0 {count=1, lds_addr,
// global_addr[56:0], type=2}; group1 {data_size=3 (8B), tensor_dim0,
// tile_dim0, tensor_dim0_stride}.  All operands block-uniform -> SGPRs.
__device__ __forceinline__ void tdm_load_1d(unsigned lds_off,
                                            unsigned long long gaddr,
                                            unsigned nElem8) {
  u32x4 g0;
  g0.x = 1u;                                                    // count=1
  g0.y = lds_off;                                               // LDS byte addr
  g0.z = (unsigned)gaddr;                                       // gaddr[31:0]
  g0.w = (unsigned)((gaddr >> 32) & 0x01FFFFFFull) | (2u << 30);// hi + type=2
  u32x8 g1;
  g1.s0 = (3u << 16);                      // workgroup_mask=0, data_size=8B
  g1.s1 = (nElem8 & 0xFFFFu) << 16;        // tensor_dim0[15:0]
  g1.s2 = (nElem8 >> 16);                  // tensor_dim0[31:16]
  g1.s3 = (nElem8 & 0xFFFFu) << 16;        // tile_dim0 (fits 16 bits)
  g1.s4 = 0u;                              // tile_dim1/2 unused
  g1.s5 = nElem8;                          // tensor_dim0_stride low
  g1.s6 = 0u;
  g1.s7 = 0u;
  asm volatile("tensor_load_to_lds %0, %1" :: "s"(g0), "s"(g1) : "memory");
}

// ---------------------------------------------------------------------------
__global__ void __launch_bounds__(256) k_norm(const float* __restrict__ x,
                                              const float* __restrict__ aw,
                                              const float* __restrict__ ab,
                                              _Float16* __restrict__ Fh,
                                              float* __restrict__ meanArr,
                                              float* __restrict__ stdArr) {
  int bc = blockIdx.x, b = bc / C_, c = bc % C_;
  const float* xp = x + (size_t)b * L_ * C_ + c;
  __shared__ float red[256];
  float s0 = 0.f, s1 = 0.f;
  for (int t = threadIdx.x; t < L_; t += 256) {
    float v = xp[(size_t)t * C_]; s0 += v; s1 += v * v;
  }
  red[threadIdx.x] = s0; __syncthreads();
  for (int o = 128; o > 0; o >>= 1) {
    if (threadIdx.x < o) red[threadIdx.x] += red[threadIdx.x + o];
    __syncthreads();
  }
  float mean = red[0] * (1.f / L_); __syncthreads();
  red[threadIdx.x] = s1; __syncthreads();
  for (int o = 128; o > 0; o >>= 1) {
    if (threadIdx.x < o) red[threadIdx.x] += red[threadIdx.x + o];
    __syncthreads();
  }
  float var = red[0] * (1.f / L_) - mean * mean;
  float sd  = sqrtf(var + 1e-5f);
  if (threadIdx.x == 0) { meanArr[bc] = mean; stdArr[bc] = sd; }
  float w = aw[c], bb = ab[c];
  for (int t = threadIdx.x; t < L_; t += 256) {
    float v = (xp[(size_t)t * C_] - mean) / sd * w + bb;
    Fh[(size_t)bc * L_ + t] = (_Float16)v;
  }
}

// ---------------------------------------------------------------------------
__global__ void __launch_bounds__(256) k_ebar(const float* __restrict__ ev,
                                              float* __restrict__ ebar) {
  int s = blockIdx.x, i = threadIdx.x;
  const float* p = ev + (size_t)s * L_ * NH + i;
  float acc = 0.f;
  for (int l = 0; l < L_; ++l) acc += p[(size_t)l * NH];
  ebar[s * NH + i] = acc * (1.f / L_);
}

// ---------------------------------------------------------------------------
__global__ void __launch_bounds__(256) k_wconv(const float* __restrict__ wr,
                                               const float* __restrict__ wi,
                                               _Float16* __restrict__ Whr,
                                               _Float16* __restrict__ Whi) {
  size_t idx = (size_t)blockIdx.x * 256 + threadIdx.x;   // [s][k][o][i]
  size_t total = (size_t)NSC * MODES * NH * NH;
  if (idx >= total) return;
  int i = (int)(idx % NH); size_t r = idx / NH;
  int o = (int)(r % NH);   r /= NH;
  int k = (int)(r % MODES);
  int s = (int)(r / MODES);
  size_t src = (((size_t)s * NH + i) * NH + o) * MODES + k;
  Whr[idx] = (_Float16)wr[src];
  Whi[idx] = (_Float16)wi[src];
}

// ---------------------------------------------------------------------------
// One workgroup per scale: sequential sweep v_{d+1} = A v_d, v_0 = B.
// H_k[i] += e^{-i w_k d} v_d[i];  G[tau=L-1-d][i][k] = e^{i w_k (d+1)} H_k[i].
__global__ void __launch_bounds__(256, 1) k_precompute(const float* __restrict__ Amats,
                                                       const float* __restrict__ Bvecs,
                                                       _Float16* __restrict__ Gt) {
  int s = blockIdx.x, i = threadIdx.x;
  const float* A = Amats + (size_t)s * NH * NH;
  _Float16* G = Gt + (size_t)s * NCOL * L_;
  __shared__ float vbuf[NH];
  float v = Bvecs[s * NH + i];
  float Hr[MODES], Hi[MODES], pr[MODES], pi[MODES], qr[MODES], qi[MODES];
  float cw[MODES], sw[MODES];
#pragma unroll
  for (int k = 0; k < MODES; ++k) {
    float w = 6.2831853071795864f * (float)k * (1.f / L_);
    cw[k] = cosf(w); sw[k] = sinf(w);
    Hr[k] = 0.f; Hi[k] = 0.f;
    pr[k] = 1.f; pi[k] = 0.f;          // e^{-i w k d}, d = 0
    qr[k] = cw[k]; qi[k] = sw[k];      // e^{+i w k (d+1)}, d = 0
  }
  for (int d = 0; d < L_; ++d) {
    __syncthreads(); vbuf[i] = v; __syncthreads();
    int tau = L_ - 1 - d;
#pragma unroll
    for (int k = 0; k < MODES; ++k) {
      Hr[k] += pr[k] * v; Hi[k] += pi[k] * v;
      float gre = qr[k] * Hr[k] - qi[k] * Hi[k];
      float gim = qr[k] * Hi[k] + qi[k] * Hr[k];
      G[(size_t)(i * 64 + 2 * k)     * L_ + tau] = (_Float16)gre;
      G[(size_t)(i * 64 + 2 * k + 1) * L_ + tau] = (_Float16)gim;
      float npr = pr[k] * cw[k] + pi[k] * sw[k];   // * e^{-i w}
      float npi = pi[k] * cw[k] - pr[k] * sw[k];
      pr[k] = npr; pi[k] = npi;
      float nqr = qr[k] * cw[k] - qi[k] * sw[k];   // * e^{+i w}
      float nqi = qi[k] * cw[k] + qr[k] * sw[k];
      qr[k] = nqr; qi[k] = nqi;
    }
    const float* Ar = A + (size_t)i * NH;
    float acc = 0.f;
#pragma unroll 8
    for (int j = 0; j < NH; ++j) acc += Ar[j] * vbuf[j];
    v = acc;
  }
}

// ---------------------------------------------------------------------------
// GEMM1: X(192 x 16384) = Fh(192 x 1024) @ G(1024 x 16384), per scale.
// 8 waves/block; 16x1024 A panel (contiguous 32KB) DMA'd to LDS via
// global_load_async_to_lds_b128.  Each wave owns 4 N-tiles (A-fragment
// reused 4x per K-step -> 4 WMMAs per iteration).
__global__ void __launch_bounds__(256, 2) k_gemm1(const _Float16* __restrict__ Fh,
                                                  const _Float16* __restrict__ Gt,
                                                  _Float16* __restrict__ Xr) {
  int mt = blockIdx.x;           // 0..11  M tile
  int ng = blockIdx.y;           // 0..31  N group (8 waves * 4 tiles)
  int s  = blockIdx.z;
  const _Float16* G = Gt + (size_t)s * NCOL * L_;
  __shared__ __align__(16) _Float16 Ah[16 * L_];   // 32 KB A panel

  {
    const _Float16* src = Fh + (size_t)mt * 16 * L_;
    unsigned ldsBase = (unsigned)(unsigned long long)(uintptr_t)&Ah[0];
    unsigned long long g0 = (unsigned long long)(uintptr_t)src;
#pragma unroll
    for (int j = 0; j < 8; ++j) {
      unsigned off = threadIdx.x * 128u + (unsigned)j * 16u;
      unsigned lo = ldsBase + off;
      unsigned long long ga = g0 + off;
      asm volatile("global_load_async_to_lds_b128 %0, %1, off"
                   :: "v"(lo), "v"(ga) : "memory");
    }
    asm volatile("s_wait_asynccnt 0" ::: "memory");
  }
  __syncthreads();

  int wave = threadIdx.x >> 5, lane = threadIdx.x & 31;
  int nt0 = (ng * 8 + wave) * 4;       // first of 4 N tiles
  int ln = lane & 15, g = lane >> 4;
  v8f acc[4] = {{}, {}, {}, {}};
  const unsigned* AhU = (const unsigned*)&Ah[0];
  for (int kt = 0; kt < L_ / 32; ++kt) {
    v16h a;
#pragma unroll
    for (int v = 0; v < 8; ++v)
      put2(a, 2 * v, AhU[(ln * L_ + kA0(v, g) + kt * 32) >> 1]);
#pragma unroll
    for (int j = 0; j < 4; ++j) {
      v16h b;
#pragma unroll
      for (int v = 0; v < 8; ++v) {
        const unsigned* bp = (const unsigned*)
            (G + ((size_t)((nt0 + j) * 16 + ln) * L_ + kB0(v, g) + kt * 32));
        put2(b, 2 * v, *bp);
      }
      acc[j] = wmma16(a, b, acc[j]);
    }
  }
#pragma unroll
  for (int j = 0; j < 4; ++j) {
    int col = (nt0 + j) * 16 + ln;
    int ii = col >> 6, kk = col & 63;
#pragma unroll
    for (int r = 0; r < 8; ++r) {
      int row = mt * 16 + r + 8 * g;
      Xr[(((size_t)s * 64 + kk) * BC + row) * NH + ii] = (_Float16)acc[j][r];
    }
  }
}

// ---------------------------------------------------------------------------
// GEMM2: for each mode k, complex (192x256)@(256x256) via 4 real WMMAs,
// fused with the irfft-at-(L-1) phase reduction.  The 8KB-contiguous B
// slices Wh*[s][k][nt*16..+15][:] are DMA'd by the TDM into LDS, double
// buffered: issue k+1, s_wait_tensorcnt, compute k.
__global__ void __launch_bounds__(32) k_gemm2(const _Float16* __restrict__ Xr,
                                              const _Float16* __restrict__ Whr,
                                              const _Float16* __restrict__ Whi,
                                              float* __restrict__ xdec) {
  int mt = blockIdx.x;           // 0..11
  int nt = blockIdx.y;           // 0..15
  int s  = blockIdx.z;
  int lane = threadIdx.x & 31, ln = lane & 15, g = lane >> 4;
  __shared__ __align__(16) _Float16 Bb[2][2][16 * NH];   // [buf][re/im] 32KB

  auto issue = [&](int k, int buf) {
    size_t off = (((size_t)s * MODES + k) * NH + nt * 16) * NH;  // halfs
    tdm_load_1d((unsigned)(uintptr_t)&Bb[buf][0][0],
                (unsigned long long)(uintptr_t)(Whr + off), 1024u);
    tdm_load_1d((unsigned)(uintptr_t)&Bb[buf][1][0],
                (unsigned long long)(uintptr_t)(Whi + off), 1024u);
  };

  issue(0, 0);
  v8f res = {};
  for (int k = 0; k < MODES; ++k) {
    if (k + 1 < MODES) {
      issue(k + 1, (k + 1) & 1);
      __builtin_amdgcn_s_wait_tensorcnt(2);   // k's pair has landed
    } else {
      __builtin_amdgcn_s_wait_tensorcnt(0);
    }
    const unsigned* bre = (const unsigned*)&Bb[k & 1][0][0];
    const unsigned* bim = (const unsigned*)&Bb[k & 1][1][0];
    const _Float16* Are = Xr + ((size_t)s * 64 + 2 * k)     * BC * NH;
    const _Float16* Aim = Xr + ((size_t)s * 64 + 2 * k + 1) * BC * NH;
    v8f a1 = {}, a2 = {}, a3 = {}, a4 = {};
    for (int it = 0; it < NH / 32; ++it) {
      v16h xre, xim, wre, wim;
#pragma unroll
      for (int v = 0; v < 8; ++v) {
        int K0 = kA0(v, g) + it * 32;
        size_t ao = (size_t)(mt * 16 + ln) * NH + K0;
        put2(xre, 2 * v, *(const unsigned*)(Are + ao));
        put2(xim, 2 * v, *(const unsigned*)(Aim + ao));
        int KB0 = kB0(v, g) + it * 32;
        unsigned bo = (unsigned)(ln * NH + KB0) >> 1;
        put2(wre, 2 * v, bre[bo]);
        put2(wim, 2 * v, bim[bo]);
      }
      a1 = wmma16(xre, wre, a1);
      a2 = wmma16(xim, wim, a2);
      a3 = wmma16(xre, wim, a3);
      a4 = wmma16(xim, wre, a4);
    }
    float th = 6.2831853071795864f * (float)k * (1.f / L_);
    float gr = (k == 0 ? 1.0f : 2.0f * cosf(th)) * (1.f / L_);
    float gs = (k == 0 ? 0.0f : 2.0f * sinf(th)) * (1.f / L_);
#pragma unroll
    for (int r = 0; r < 8; ++r)
      res[r] += gr * (a1[r] - a2[r]) + gs * (a3[r] + a4[r]);
  }
#pragma unroll
  for (int r = 0; r < 8; ++r) {
    int row = mt * 16 + r + 8 * g;
    int o = nt * 16 + ln;
    xdec[((size_t)s * BC + row) * NH + o] = res[r];
  }
}

// ---------------------------------------------------------------------------
__global__ void __launch_bounds__(256) k_final(const float* __restrict__ xdec,
                                               const float* __restrict__ ebar,
                                               const float* __restrict__ wmlp,
                                               const float* __restrict__ bmlp,
                                               const float* __restrict__ aw,
                                               const float* __restrict__ ab,
                                               const float* __restrict__ meanArr,
                                               const float* __restrict__ stdArr,
                                               const float* __restrict__ wfc,
                                               const float* __restrict__ bfc,
                                               float* __restrict__ out) {
  __shared__ float feat[BC];
  int t = threadIdx.x;
  if (t < BC) {
    float xdm = bmlp[0];
    for (int s = 0; s < NSC; ++s) {
      const float* xp = xdec + ((size_t)s * BC + t) * NH;
      const float* ep = ebar + s * NH;
      float acc = 0.f;
      for (int o = 0; o < NH; ++o) acc += xp[o] * ep[o];
      xdm += wmlp[s] * acc;
    }
    int c = t % C_;
    feat[t] = (xdm - ab[c]) / (aw[c] + 1e-10f) * stdArr[t] + meanArr[t];
  }
  __syncthreads();
  if (t < B_ * OUTD) {
    int b = t / OUTD, od = t % OUTD;
    float acc = bfc[od];
    for (int c = 0; c < C_; ++c) acc += feat[b * C_ + c] * wfc[od * C_ + c];
    out[t] = acc;
  }
}

// ---------------------------------------------------------------------------
extern "C" void kernel_launch(void* const* d_in, const int* in_sizes, int n_in,
                              void* d_out, int out_size, void* d_ws, size_t ws_size,
                              hipStream_t stream) {
  (void)in_sizes; (void)n_in; (void)out_size; (void)ws_size;
  const float* x_enc = (const float*)d_in[0];
  const float* aw    = (const float*)d_in[1];
  const float* ab    = (const float*)d_in[2];
  const float* wsr   = (const float*)d_in[3];
  const float* wsi   = (const float*)d_in[4];
  const float* wmlp  = (const float*)d_in[5];
  const float* bmlp  = (const float*)d_in[6];
  const float* wfc   = (const float*)d_in[7];
  const float* bfc   = (const float*)d_in[8];
  const float* Amats = (const float*)d_in[9];
  const float* Bvecs = (const float*)d_in[10];
  const float* evl   = (const float*)d_in[11];
  float* out = (float*)d_out;

  char* w = (char*)d_ws;
  auto alloc = [&](size_t bytes) {
    char* p = w; w += (bytes + 255) & ~(size_t)255; return p;
  };
  _Float16* Fh   = (_Float16*)alloc((size_t)BC * L_ * 2);
  float* meanArr = (float*)alloc((size_t)BC * 4);
  float* stdArr  = (float*)alloc((size_t)BC * 4);
  float* ebar    = (float*)alloc((size_t)NSC * NH * 4);
  _Float16* Gt   = (_Float16*)alloc((size_t)NSC * NCOL * L_ * 2);   // 96 MB
  _Float16* Xr   = (_Float16*)alloc((size_t)NSC * 64 * BC * NH * 2);
  _Float16* Whr  = (_Float16*)alloc((size_t)NSC * MODES * NH * NH * 2);
  _Float16* Whi  = (_Float16*)alloc((size_t)NSC * MODES * NH * NH * 2);
  float* xdec    = (float*)alloc((size_t)NSC * BC * NH * 4);

  k_norm<<<dim3(BC), dim3(256), 0, stream>>>(x_enc, aw, ab, Fh, meanArr, stdArr);
  k_ebar<<<dim3(NSC), dim3(NH), 0, stream>>>(evl, ebar);
  size_t wtot = (size_t)NSC * MODES * NH * NH;
  k_wconv<<<dim3((unsigned)((wtot + 255) / 256)), dim3(256), 0, stream>>>(wsr, wsi, Whr, Whi);
  k_precompute<<<dim3(NSC), dim3(NH), 0, stream>>>(Amats, Bvecs, Gt);
  k_gemm1<<<dim3(12, 32, NSC), dim3(256), 0, stream>>>(Fh, Gt, Xr);
  k_gemm2<<<dim3(12, 16, NSC), dim3(32), 0, stream>>>(Xr, Whr, Whi, xdec);
  k_final<<<dim3(1), dim3(256), 0, stream>>>(xdec, ebar, wmlp, bmlp, aw, ab,
                                             meanArr, stdArr, wfc, bfc, out);
}
